// GCN_sim_23562190586236
// MI455X (gfx1250) — matrix operand order repaired and checked by reading, not compiled, not run on hardware
//
#include <hip/hip_runtime.h>

typedef unsigned short u16;
typedef unsigned int   u32;
typedef __attribute__((ext_vector_type(16))) __bf16 bf16x16;
typedef __attribute__((ext_vector_type(8)))  float  floatx8;

#define NROWS 8192
#define D1    1024
#define DF    256
#define TOPK  32

struct alignas(16) U128 { u32 x, y, z, w; };
struct Frag32 { U128 a, b; };                 // 32 bytes == 16 bf16
struct alignas(8) U16x4 { u16 s[4]; };

static_assert(sizeof(Frag32) == sizeof(bf16x16), "frag size");

#define FZERO {0.f, 0.f, 0.f, 0.f, 0.f, 0.f, 0.f, 0.f}

__device__ __forceinline__ u16 f2bf(float f) {
  u32 u = __float_as_uint(f);
  u32 r = u + 0x7FFFu + ((u >> 16) & 1u);      // round-to-nearest-even
  return (u16)(r >> 16);
}

// A fragment (16x32 bf16, row-major source). p already points at
// row*ld + kb + ((lane>>4)*8). Elements 0..7 = p[0..7], 8..15 = p[16..23].
__device__ __forceinline__ bf16x16 load_a_frag(const u16* p) {
  Frag32 t;
  t.a = *(const U128*)(p);
  t.b = *(const U128*)(p + 16);
  return __builtin_bit_cast(bf16x16, t);
}
// B fragment (32x16 bf16) loaded from B^T row-major. p points at
// n_row*ld + kb + ((lane>>4)*8). Elements 0..7 = p[0..7], 8..15 = p[16..23].
__device__ __forceinline__ bf16x16 load_b_frag(const u16* p) {
  Frag32 t;
  t.a = *(const U128*)(p);
  t.b = *(const U128*)(p + 16);
  return __builtin_bit_cast(bf16x16, t);
}

__device__ __forceinline__ floatx8 wmma_bf16(bf16x16 a, bf16x16 b, floatx8 c) {
  return __builtin_amdgcn_wmma_f32_16x16x32_bf16(false, a, false, b, (short)0, c,
                                                 false, false);
}

// ---------------- weight prep ----------------
__global__ __launch_bounds__(256) void cvt_bf16_kernel(const float* __restrict__ src,
                                                       u16* __restrict__ dst, int n) {
  int i = blockIdx.x * 256 + threadIdx.x;
  if (i < n) dst[i] = f2bf(src[i]);
}

// gc_w is [K=D1][N=D1]; build B^T layout: dst[n*D1+k] = src[k*D1+n]
__global__ __launch_bounds__(256) void transpose_bf16_kernel(const float* __restrict__ src,
                                                             u16* __restrict__ dst) {
  int i = blockIdx.x * 256 + threadIdx.x;   // over D1*D1
  int n = i >> 10, k = i & (D1 - 1);
  dst[i] = f2bf(src[(size_t)k * D1 + n]);
}

// ---------------- layernorm -> bf16 ----------------
__global__ __launch_bounds__(256) void ln_kernel(const float* __restrict__ X,
                                                 const float* __restrict__ gamma,
                                                 const float* __restrict__ beta,
                                                 u16* __restrict__ X1h) {
  const int row = blockIdx.x, tid = threadIdx.x;
  const float4 v = ((const float4*)(X + (size_t)row * D1))[tid];
  float s  = v.x + v.y + v.z + v.w;
  float ss = v.x * v.x + v.y * v.y + v.z * v.z + v.w * v.w;
  __shared__ float rs[256], rq[256];
  rs[tid] = s; rq[tid] = ss;
  __syncthreads();
  for (int st = 128; st > 0; st >>= 1) {
    if (tid < st) { rs[tid] += rs[tid + st]; rq[tid] += rq[tid + st]; }
    __syncthreads();
  }
  const float mean = rs[0] * (1.0f / D1);
  const float var  = rq[0] * (1.0f / D1) - mean * mean;
  const float rstd = rsqrtf(var + 1e-5f);
  const float4 g = ((const float4*)gamma)[tid];
  const float4 b = ((const float4*)beta)[tid];
  U16x4 o;
  o.s[0] = f2bf((v.x - mean) * rstd * g.x + b.x);
  o.s[1] = f2bf((v.y - mean) * rstd * g.y + b.y);
  o.s[2] = f2bf((v.z - mean) * rstd * g.z + b.z);
  o.s[3] = f2bf((v.w - mean) * rstd * g.w + b.w);
  ((U16x4*)(X1h + (size_t)row * D1))[tid] = o;
}

// ---------------- bf16 WMMA GEMM, 2x2 register blocking per wave ----------------
// C[M][Nc] = A[M][K] * Bt[Nc][K]^T (+bias). Block tile 64x128, wave tile 32x32.
__global__ __launch_bounds__(256) void gemm_bf16_kernel(const u16* __restrict__ A,
                                                        const u16* __restrict__ Bt,
                                                        const float* __restrict__ bias,
                                                        float* __restrict__ C,
                                                        int K, int Nc) {
  const int tid  = threadIdx.x;
  const int lane = tid & 31, wave = tid >> 5;
  const int mw = wave >> 2, nw = wave & 3;           // 2 x 4 wave grid
  const int m0 = blockIdx.x * 64 + mw * 32;
  const int n0 = blockIdx.y * 128 + nw * 32;
  const int l15 = lane & 15, hi = lane >> 4;

  const u16* ap0 = A  + (size_t)(m0 + l15) * K + hi * 8;
  const u16* ap1 = ap0 + (size_t)16 * K;
  const u16* bp0 = Bt + (size_t)(n0 + l15) * K + hi * 8;
  const u16* bp1 = bp0 + (size_t)16 * K;

  floatx8 acc00 = FZERO, acc01 = FZERO, acc10 = FZERO, acc11 = FZERO;
  for (int kb = 0; kb < K; kb += 32) {
    // stream the K dimension ahead (global_prefetch_b8)
    __builtin_prefetch(ap0 + kb + 512, 0, 1);
    __builtin_prefetch(ap1 + kb + 512, 0, 1);
    __builtin_prefetch(bp0 + kb + 512, 0, 1);
    __builtin_prefetch(bp1 + kb + 512, 0, 1);
    bf16x16 a0 = load_a_frag(ap0 + kb);
    bf16x16 a1 = load_a_frag(ap1 + kb);
    bf16x16 b0 = load_b_frag(bp0 + kb);
    bf16x16 b1 = load_b_frag(bp1 + kb);
    acc00 = wmma_bf16(a0, b0, acc00);
    acc01 = wmma_bf16(a0, b1, acc01);
    acc10 = wmma_bf16(a1, b0, acc10);
    acc11 = wmma_bf16(a1, b1, acc11);
  }
  const int nA = n0 + l15, nB = n0 + 16 + l15;
  const float bvA = bias ? bias[nA] : 0.0f;
  const float bvB = bias ? bias[nB] : 0.0f;
#pragma unroll
  for (int r = 0; r < 8; ++r) {
    const int mA = m0 + r + hi * 8;
    const int mB = mA + 16;
    C[(size_t)mA * Nc + nA] = acc00[r] + bvA;
    C[(size_t)mA * Nc + nB] = acc01[r] + bvB;
    C[(size_t)mB * Nc + nA] = acc10[r] + bvA;
    C[(size_t)mB * Nc + nB] = acc11[r] + bvB;
  }
}

// ---------------- row L2-normalize -> bf16 ----------------
__global__ __launch_bounds__(256) void rownorm_kernel(const float* __restrict__ Xf,
                                                      u16* __restrict__ Xnh) {
  const int row = blockIdx.x, tid = threadIdx.x;
  const float v = Xf[(size_t)row * DF + tid];
  __shared__ float rq[256];
  rq[tid] = v * v;
  __syncthreads();
  for (int st = 128; st > 0; st >>= 1) {
    if (tid < st) rq[tid] += rq[tid + st];
    __syncthreads();
  }
  const float rinv = rsqrtf(rq[0]);
  Xnh[(size_t)row * DF + tid] = f2bf(v * rinv);
}

// Cooperative wave-wide top-K maintenance over one row's 256 fresh scores.
// All 32 lanes execute; mn/mnp are wave-uniform. Insertions are rare, so the
// common path is 8 x (LDS read + compare + ballot).
__device__ __forceinline__ void topk_scan(float* __restrict__ tv, int* __restrict__ ti,
                                          const float* __restrict__ sc, int j0,
                                          float& mn, int& mnp, int lane) {
#pragma unroll 1
  for (int g = 0; g < 256; g += 32) {
    const float v = sc[g + lane];
    unsigned mm = (unsigned)__ballot(v > mn);
    while (mm) {
      const int src = __builtin_ctz(mm);
      mm &= mm - 1;
      const float vv = __shfl(v, src, 32);
      if (vv > mn) {                       // min may have risen since ballot
        tv[mnp] = vv;                      // all lanes: same address, same value
        ti[mnp] = j0 + g + src;
        float cv = tv[lane];
        int   cp = lane;
#pragma unroll
        for (int off = 16; off; off >>= 1) {
          const float ov = __shfl_xor(cv, off, 32);
          const int   op = __shfl_xor(cp, off, 32);
          if (ov < cv) { cv = ov; cp = op; }
        }
        mn = cv; mnp = cp;                 // wave-uniform
      }
    }
  }
}

// ---------------- fused similarity + top-32 + softmax ----------------
// Block: 16 rows, 8 waves. Each wave: 2 accumulators (32 cols of a 256-col
// j-tile) for the WMMA phase, then owns 2 rows for the ballot-based top-k.
__global__ __launch_bounds__(256) void sim_topk_kernel(const u16* __restrict__ Xnh,
                                                       float* __restrict__ topw,
                                                       int* __restrict__ topidx) {
  const int tid  = threadIdx.x;
  const int lane = tid & 31, wave = tid >> 5;
  const int m0 = blockIdx.x * 16;
  const int l15 = lane & 15, hi = lane >> 4;

  __shared__ float s_sc[16][260];       // 16 rows x 256 cols (+pad)
  __shared__ float s_tv[16][TOPK];
  __shared__ int   s_ti[16][TOPK];

  for (int q = tid; q < 16 * TOPK; q += 256) {
    s_tv[q >> 5][q & 31] = -__builtin_inff();
    s_ti[q >> 5][q & 31] = 0;
  }
  __syncthreads();

  // preload the 16x256 A tile as 8 fragments (K=256)
  bf16x16 af[8];
  {
    const u16* ap = Xnh + (size_t)(m0 + l15) * DF + hi * 8;
#pragma unroll
    for (int k = 0; k < 8; ++k) af[k] = load_a_frag(ap + k * 32);
  }

  const int r0 = wave * 2, r1 = r0 + 1;
  float mn0 = -__builtin_inff(), mn1 = -__builtin_inff();
  int   mp0 = 0, mp1 = 0;

  for (int j0 = 0; j0 < NROWS; j0 += 256) {
    const u16* bp0 = Xnh + (size_t)(j0 + wave * 32 + l15) * DF + hi * 8;
    const u16* bp1 = bp0 + (size_t)16 * DF;
    floatx8 c0 = FZERO, c1 = FZERO;
#pragma unroll
    for (int k = 0; k < 8; ++k) {
      bf16x16 b0 = load_b_frag(bp0 + k * 32);
      bf16x16 b1 = load_b_frag(bp1 + k * 32);
      c0 = wmma_bf16(af[k], b0, c0);
      c1 = wmma_bf16(af[k], b1, c1);
    }
#pragma unroll
    for (int r = 0; r < 8; ++r) {
      s_sc[r + hi * 8][wave * 32 + l15]      = c0[r];
      s_sc[r + hi * 8][wave * 32 + 16 + l15] = c1[r];
    }
    __syncthreads();
    topk_scan(s_tv[r0], s_ti[r0], s_sc[r0], j0, mn0, mp0, lane);
    topk_scan(s_tv[r1], s_ti[r1], s_sc[r1], j0, mn1, mp1, lane);
    __syncthreads();
  }

  // softmax over the 32 survivors of each row: one lane per entry
#pragma unroll
  for (int rr = 0; rr < 2; ++rr) {
    const int r = wave * 2 + rr;
    const int row = m0 + r;
    const float v = s_tv[r][lane];
    float mx = v;
#pragma unroll
    for (int off = 16; off; off >>= 1) mx = fmaxf(mx, __shfl_xor(mx, off, 32));
    const float e = __expf(v - mx);
    float s = e;
#pragma unroll
    for (int off = 16; off; off >>= 1) s += __shfl_xor(s, off, 32);
    topw[(size_t)row * TOPK + lane]   = e / s;
    topidx[(size_t)row * TOPK + lane] = s_ti[r][lane];
  }
}

// ---------------- sparse aggregate + bias + residual ----------------
__global__ __launch_bounds__(256) void gather_out_kernel(const float* __restrict__ X,
                                                         const float* __restrict__ gcb,
                                                         const float* __restrict__ H,
                                                         const float* __restrict__ topw,
                                                         const int* __restrict__ topidx,
                                                         float* __restrict__ out) {
  const int row = blockIdx.x, tid = threadIdx.x;
  __shared__ float w[TOPK];
  __shared__ int   idx[TOPK];
  if (tid < TOPK) {
    w[tid]   = topw[(size_t)row * TOPK + tid];
    idx[tid] = topidx[(size_t)row * TOPK + tid];
  }
  __syncthreads();
  float acc0 = 0.f, acc1 = 0.f, acc2 = 0.f, acc3 = 0.f;
  for (int t = 0; t < TOPK; ++t) {
    const float* hr = H + (size_t)idx[t] * D1;
    const float wt = w[t];
    acc0 += wt * hr[tid];
    acc1 += wt * hr[tid + 256];
    acc2 += wt * hr[tid + 512];
    acc3 += wt * hr[tid + 768];
  }
  const size_t base = (size_t)row * D1;
  out[base + tid      ] = X[base + tid      ] + acc0 + gcb[tid];
  out[base + tid + 256] = X[base + tid + 256] + acc1 + gcb[tid + 256];
  out[base + tid + 512] = X[base + tid + 512] + acc2 + gcb[tid + 512];
  out[base + tid + 768] = X[base + tid + 768] + acc3 + gcb[tid + 768];
}

extern "C" void kernel_launch(void* const* d_in, const int* in_sizes, int n_in,
                              void* d_out, int out_size, void* d_ws, size_t ws_size,
                              hipStream_t stream) {
  const float* X     = (const float*)d_in[0];
  const float* gam   = (const float*)d_in[1];
  const float* bet   = (const float*)d_in[2];
  const float* fcw   = (const float*)d_in[3];
  const float* fcb   = (const float*)d_in[4];
  const float* gcw   = (const float*)d_in[5];
  const float* gcb   = (const float*)d_in[6];
  float* out = (float*)d_out;

  size_t off = 0;
  auto carve = [&](size_t bytes) -> void* {
    void* p = (char*)d_ws + off;
    off += (bytes + 255) & ~(size_t)255;
    return p;
  };
  u16*   X1h   = (u16*)  carve((size_t)NROWS * D1 * 2);   // 16 MB
  u16*   fcwh  = (u16*)  carve((size_t)DF * D1 * 2);      // 0.5 MB
  u16*   gcwh  = (u16*)  carve((size_t)D1 * D1 * 2);      // 2 MB
  float* Xf    = (float*)carve((size_t)NROWS * DF * 4);   // 8 MB
  u16*   Xnh   = (u16*)  carve((size_t)NROWS * DF * 2);   // 4 MB
  float* H     = (float*)carve((size_t)NROWS * D1 * 4);   // 32 MB
  float* topw  = (float*)carve((size_t)NROWS * TOPK * 4); // 1 MB
  int*   topix = (int*)  carve((size_t)NROWS * TOPK * 4); // 1 MB

  // weight conversion
  cvt_bf16_kernel<<<(DF * D1 + 255) / 256, 256, 0, stream>>>(fcw, fcwh, DF * D1);
  transpose_bf16_kernel<<<(D1 * D1) / 256, 256, 0, stream>>>(gcw, gcwh);

  // layernorm
  ln_kernel<<<NROWS, 256, 0, stream>>>(X, gam, bet, X1h);

  // Xf = X1 @ fc_w^T + fc_b   (M=8192, N=256, K=1024)
  gemm_bf16_kernel<<<dim3(NROWS / 64, DF / 128), 256, 0, stream>>>(X1h, fcwh, fcb, Xf,
                                                                   D1, DF);
  // row-normalize -> bf16
  rownorm_kernel<<<NROWS, 256, 0, stream>>>(Xf, Xnh);

  // H = X1 @ gc_w             (M=8192, N=1024, K=1024)
  gemm_bf16_kernel<<<dim3(NROWS / 64, D1 / 128), 256, 0, stream>>>(X1h, gcwh, nullptr,
                                                                   H, D1, D1);
  // similarity + top-32 + softmax (fused, never materializes 8192x8192)
  sim_topk_kernel<<<NROWS / 16, 256, 0, stream>>>(Xnh, topw, topix);

  // out = X + G @ H + gc_b
  gather_out_kernel<<<NROWS, 256, 0, stream>>>(X, gcb, H, topw, topix, out);

  (void)in_sizes; (void)n_in; (void)out_size; (void)ws_size;
}
